// AttentionModule_16398185136487
// MI455X (gfx1250) — matrix operand order repaired and checked by reading, not compiled, not run on hardware
//
#include <hip/hip_runtime.h>
#include <math.h>

// ---------------------------------------------------------------------------
// MI455X (gfx1250) implementation.
//   - All GEMMs on v_wmma_f32_16x16x32_bf16 (f32 accumulate).
//   - Q stored as WMMA A-fragments, K/V as WMMA B-fragments (contiguous k = 16*half + s
//     layouts) so the 137-GFLOP gate matmul is pure load + wmma.
//   - Gate kernel: 4 waves x 2 query tiles per block share one K stream,
//     staged through double-buffered LDS with GLOBAL_LOAD_ASYNC_TO_LDS_B128
//     (ASYNCcnt) so the next tile's fill overlaps the current tile's WMMAs.
//   - final = concat([gate_ba*VPa, gate_aa*VPa, gate_ab*VPb, gate_bb*VPb])+b128
//     (the reference's trailing softmax is over a size-1 axis == identity).
// ---------------------------------------------------------------------------

typedef __attribute__((ext_vector_type(16))) __bf16          v16bf;
typedef __attribute__((ext_vector_type(16))) unsigned short  v16u;
typedef __attribute__((ext_vector_type(8)))  float           v8f;

#define WMMA_BF16(a, b, c) \
  __builtin_amdgcn_wmma_f32_16x16x32_bf16(false, (a), false, (b), (short)0, (c), false, false)

static constexpr size_t kMB      = 1048576;
// six 4MB bf16 fragment matrices: QA,KA,VA,QB,KB,VB  (cid 0..5)
static constexpr size_t OFF_VP   = 24 * kMB;                 // VPa (4MB), VPb (4MB), f32
static constexpr size_t OFF_M    = 32 * kMB;                 // m logits: 4 gates * 8192 f32
static constexpr size_t OFF_G    = OFF_M + 131072;           // gate weights, same shape
static constexpr size_t OFF_WF   = OFF_G + 131072;           // wq/wk/wv A-frags, 128KB each
static constexpr size_t OFF_W128 = OFF_WF + 3 * 131072;      // w128 A-frags, 64KB

__device__ __forceinline__ unsigned short f2b(float f) {     // f32 -> bf16 RNE
  unsigned u = __builtin_bit_cast(unsigned, f);
  u += 0x7FFFu + ((u >> 16) & 1u);
  return (unsigned short)(u >> 16);
}

// byte offset of element (mm,kk) inside one 1KB A-fragment (16x32 bf16)
__device__ __forceinline__ int a_elem(int mm, int kk) {
  int hi = kk >> 4, r2 = kk & 15;
  return ((mm + (hi << 4)) << 5) + (r2 << 1);
}
// byte offset of element (kk,nn) inside one 1KB B-fragment (32x16 bf16)
__device__ __forceinline__ int b_elem(int kk, int nn) {
  int hi = kk >> 4, r2 = kk & 15;
  return ((nn + (hi << 4)) << 5) + (r2 << 1);
}

// gfx1250 async global->LDS copy, GVS mode: mem = SGPR base + 32b VGPR offset.
// LDS destination byte address is carried in a VGPR (ISA 15.18.3 op 98).
__device__ __forceinline__ void async_ld_b128(const void* sbase, unsigned voff,
                                              unsigned lds_addr) {
  asm volatile("global_load_async_to_lds_b128 %0, %1, %2"
               :: "v"(lds_addr), "v"(voff), "s"(sbase)
               : "memory");
}
__device__ __forceinline__ void wait_async0() {
  asm volatile("s_wait_asynccnt 0" ::: "memory");
}
// ISA 10.2: generic LDS addresses map to LDS offset via addr[31:0].
__device__ __forceinline__ unsigned lds_off(const void* p) {
  return (unsigned)(unsigned long long)p;
}

// --------------------------- weight pre-swizzle ----------------------------
__global__ __launch_bounds__(256) void prep_weights_kernel(
    const float* __restrict__ wq, const float* __restrict__ wk,
    const float* __restrict__ wv, const float* __restrict__ w128,
    char* __restrict__ ws) {
  int idx = blockIdx.x * 256 + threadIdx.x;
  float v; size_t base; int m, k;
  if (idx < 196608) {                       // wq/wk/wv: 3 x 256x256
    int wsel = idx >> 16; int e = idx & 65535; m = e >> 8; k = e & 255;
    const float* src = (wsel == 0) ? wq : ((wsel == 1) ? wk : wv);
    v = src[e]; base = OFF_WF + (size_t)wsel * 131072;
  } else if (idx < 229376) {                // w128: 128x256
    int e = idx - 196608; m = e >> 8; k = e & 255;
    v = w128[e]; base = OFF_W128;
  } else return;
  int ot = m >> 4, mm = m & 15, kc = k >> 5, kk = k & 31;
  *(unsigned short*)(ws + base + (((size_t)((ot << 3) + kc)) << 10) + a_elem(mm, kk)) = f2b(v);
}

// ------------------- QKV convs -> swizzled bf16 fragments ------------------
// WSEL: 0=q 1=k 2=v (compile-time -> straight-line store swizzle).
// grid (64 pixel-tiles, 8 batches, 2 inputs), 256 threads (8 waves).
template <int WSEL>
__global__ __launch_bounds__(256) void conv_qkv_kernel(
    const float* __restrict__ x4, const float* __restrict__ x3,
    const float* __restrict__ bq, const float* __restrict__ bk,
    const float* __restrict__ bv, char* __restrict__ ws) {
  __shared__ __align__(32) unsigned short xlds[16 * 272];    // [pixel][cin], padded
  const int pt = blockIdx.x, b = blockIdx.y, t = blockIdx.z;
  const int cid = t * 3 + WSEL;
  const float* x    = (t == 0) ? x4 : x3;
  const float* bias = (WSEL == 0) ? bq : ((WSEL == 1) ? bk : bv);
  const char*  wfrag = ws + OFF_WF + (size_t)WSEL * 131072;
  const int tid = threadIdx.x;
  for (int i = tid; i < 4096; i += 256) {
    int c = i >> 4, n = i & 15;
    xlds[n * 272 + c] = f2b(x[((size_t)(b * 256 + c) << 10) + pt * 16 + n]);
  }
  __syncthreads();
  const int wid = tid >> 5, L = tid & 31, hi = L >> 4, n = L & 15;
  char* outbase = ws + (size_t)cid * 4 * kMB;
  for (int oti = 0; oti < 2; ++oti) {
    const int ot = wid + oti * 8;                            // output-channel tile
    v8f acc = {0.f, 0.f, 0.f, 0.f, 0.f, 0.f, 0.f, 0.f};
    #pragma unroll
    for (int kc = 0; kc < 8; ++kc) {
      v16bf a = *(const v16bf*)(wfrag + (((ot << 3) + kc) << 10) + L * 32);
      v16u  bu = *(const v16u*)(&xlds[n * 272 + kc * 32 + hi * 16]);
      acc = WMMA_BF16(a, __builtin_bit_cast(v16bf, bu), acc);
    }
    #pragma unroll
    for (int r = 0; r < 8; ++r) {
      int m   = r + (hi << 3);
      int och = ot * 16 + m;                                 // feature channel
      unsigned short us = f2b(acc[r] + bias[och]);
      int kcc = och >> 5, kk = och & 31;
      size_t fbase = ((size_t)(((b * 64 + pt) << 3) + kcc)) << 10;
      // Q -> A-fragment (row = pixel, K = channel); K/V -> B-fragment
      int eo = (WSEL == 0) ? a_elem(n, kk) : b_elem(kk, n);
      *(unsigned short*)(outbase + fbase + eo) = us;
    }
  }
}

// -------------------------- VP = w128 @ Vfeat ------------------------------
// grid (64 pixel-tiles, 8 batches, 2 inputs), 256 threads (8 waves = 128 och).
__global__ __launch_bounds__(256) void vp_conv_kernel(char* __restrict__ ws) {
  const int pt = blockIdx.x, b = blockIdx.y, t = blockIdx.z;
  const int tid = threadIdx.x, ot = tid >> 5, L = tid & 31, hi = L >> 4, n = L & 15;
  const char* vfrag = ws + (size_t)(t ? 5 : 2) * 4 * kMB;    // VA=cid2, VB=cid5
  const char* wfrag = ws + OFF_W128;
  v8f acc = {0.f, 0.f, 0.f, 0.f, 0.f, 0.f, 0.f, 0.f};
  #pragma unroll
  for (int kc = 0; kc < 8; ++kc) {
    v16bf a  = *(const v16bf*)(wfrag + (((ot << 3) + kc) << 10) + L * 32);
    v16bf bb = *(const v16bf*)(vfrag + (((size_t)(((b * 64 + pt) << 3) + kc)) << 10) + L * 32);
    acc = WMMA_BF16(a, bb, acc);
  }
  float* vp = (float*)(ws + OFF_VP + (size_t)t * 4 * kMB);
  #pragma unroll
  for (int r = 0; r < 8; ++r) {
    int och = ot * 16 + r + (hi << 3);
    vp[((size_t)((b << 7) + och) << 10) + pt * 16 + n] = acc[r];
  }
}

// ------------------ gate GEMM (8192x8192x256) + max + mean -----------------
// grid (64, 4 gates), 128 threads. 4 waves x 2 query tiles share one K stream
// through double-buffered LDS filled with async global->LDS copies.
__global__ __launch_bounds__(128) void gate_kernel(char* __restrict__ ws) {
  __shared__ __align__(32) char lbuf[2 * 8192];              // double buffer
  const int tid = threadIdx.x, wid = tid >> 5, L = tid & 31;
  const int g  = blockIdx.y;                                 // 0=aa 1=bb 2=ab 3=ba
  const int qt0 = blockIdx.x * 8 + wid * 2;                  // two query tiles/wave
  const int qsel = (g == 1 || g == 3) ? 1 : 0;               // q from x3?
  const int ksel = (g == 1 || g == 2) ? 1 : 0;               // k from x3?
  const char* qbase = ws + (size_t)(qsel * 3) * 4 * kMB;     // QA / QB
  const char* kbase = ws + (size_t)(1 + ksel * 3) * 4 * kMB; // KA / KB
  const unsigned lds0 = lds_off(&lbuf[0]);

  v16bf a0[8], a1[8];
  #pragma unroll
  for (int kc = 0; kc < 8; ++kc) {
    a0[kc] = *(const v16bf*)(qbase + (((size_t)((qt0 << 3) + kc)) << 10) + L * 32);
    a1[kc] = *(const v16bf*)(qbase + (((size_t)(((qt0 + 1) << 3) + kc)) << 10) + L * 32);
  }
  float msum0[8], msum1[8], mx0[8], mx1[8];
  #pragma unroll
  for (int r = 0; r < 8; ++r) { msum0[r] = msum1[r] = 0.f; mx0[r] = mx1[r] = -3.0e38f; }

  // prologue: stage key tile 0 into buffer 0 (each thread copies 4 x 16B)
  #pragma unroll
  for (int j = 0; j < 4; ++j) {
    unsigned o = (unsigned)(tid * 16 + j * 2048);
    async_ld_b128(kbase, o, lds0 + o);
  }

  for (int tile = 0; tile < 512; ++tile) {                   // 8 key-batches x 64 tiles
    const unsigned cur = (tile & 1) ? 8192u : 0u;
    wait_async0();
    __syncthreads();
    if (tile + 1 < 512) {                                    // overlap next fill
      const char* gp = kbase + (size_t)(tile + 1) * 8192;
      const unsigned nb = lds0 + (((tile + 1) & 1) ? 8192u : 0u);
      #pragma unroll
      for (int j = 0; j < 4; ++j) {
        unsigned o = (unsigned)(tid * 16 + j * 2048);
        async_ld_b128(gp, o, nb + o);
      }
    }
    v8f acc0 = {0.f, 0.f, 0.f, 0.f, 0.f, 0.f, 0.f, 0.f};
    v8f acc1 = {0.f, 0.f, 0.f, 0.f, 0.f, 0.f, 0.f, 0.f};
    const char* bptr = &lbuf[cur] + L * 32;
    #pragma unroll
    for (int kc = 0; kc < 8; ++kc) {
      v16bf bf = *(const v16bf*)(bptr + kc * 1024);
      acc0 = WMMA_BF16(a0[kc], bf, acc0);
      acc1 = WMMA_BF16(a1[kc], bf, acc1);
    }
    #pragma unroll
    for (int r = 0; r < 8; ++r) {
      mx0[r] = fmaxf(mx0[r], acc0[r]);
      mx1[r] = fmaxf(mx1[r], acc1[r]);
    }
    if ((tile & 63) == 63) {                                 // key-batch boundary
      #pragma unroll
      for (int r = 0; r < 8; ++r) {                          // max over 16 keys/half
        float v = mx0[r];
        v = fmaxf(v, __shfl_xor(v, 1, 32));
        v = fmaxf(v, __shfl_xor(v, 2, 32));
        v = fmaxf(v, __shfl_xor(v, 4, 32));
        v = fmaxf(v, __shfl_xor(v, 8, 32));
        msum0[r] += v; mx0[r] = -3.0e38f;
        float u = mx1[r];
        u = fmaxf(u, __shfl_xor(u, 1, 32));
        u = fmaxf(u, __shfl_xor(u, 2, 32));
        u = fmaxf(u, __shfl_xor(u, 4, 32));
        u = fmaxf(u, __shfl_xor(u, 8, 32));
        msum1[r] += u; mx1[r] = -3.0e38f;
      }
    }
  }
  if ((L & 15) == 0) {
    int hi = L >> 4;
    float* mbuf = (float*)(ws + OFF_M);
    #pragma unroll
    for (int r = 0; r < 8; ++r) {
      mbuf[g * 8192 + qt0 * 16 + r + (hi << 3)]       = msum0[r] * 0.125f; // mean/8
      mbuf[g * 8192 + (qt0 + 1) * 16 + r + (hi << 3)] = msum1[r] * 0.125f;
    }
  }
}

// ----------------------- per (gate,batch) softmax over HW ------------------
__global__ __launch_bounds__(256) void softmax_kernel(char* __restrict__ ws) {
  __shared__ float red[256];
  const int g = blockIdx.x >> 3, b = blockIdx.x & 7, tid = threadIdx.x;
  const float* mrow = (const float*)(ws + OFF_M) + (g * 8 + b) * 1024;
  float*       grow = (float*)(ws + OFF_G) + (g * 8 + b) * 1024;
  const float scale = 0.0625f;                               // 1/sqrt(256)
  float v[4], mx = -3.0e38f;
  #pragma unroll
  for (int i = 0; i < 4; ++i) { v[i] = mrow[tid + 256 * i] * scale; mx = fmaxf(mx, v[i]); }
  red[tid] = mx; __syncthreads();
  for (int s = 128; s > 0; s >>= 1) { if (tid < s) red[tid] = fmaxf(red[tid], red[tid + s]); __syncthreads(); }
  float rmax = red[0]; __syncthreads();
  float e[4], sum = 0.f;
  #pragma unroll
  for (int i = 0; i < 4; ++i) { e[i] = expf(v[i] - rmax); sum += e[i]; }
  red[tid] = sum; __syncthreads();
  for (int s = 128; s > 0; s >>= 1) { if (tid < s) red[tid] += red[tid + s]; __syncthreads(); }
  float inv = 1.0f / red[0];
  #pragma unroll
  for (int i = 0; i < 4; ++i) grow[tid + 256 * i] = e[i] * inv;
}

// -------------------- final: gate * VP + b128, concat order ----------------
__global__ __launch_bounds__(256) void final_kernel(
    const float* __restrict__ b128, const char* __restrict__ ws,
    float* __restrict__ out) {
  const int idx = blockIdx.x * 256 + threadIdx.x;            // < 8*512*1024
  const int b = idx >> 19, rem = idx & 524287;
  const int ch = rem >> 10, hw = rem & 1023;
  const int group = ch >> 7, o = ch & 127;
  const int gmap[4] = {3, 0, 2, 1};                          // [x34, a_x4, x43, b_x3]
  const int gid = gmap[group], t = group >> 1;               // t: 0->VPa, 1->VPb
  const float* gbuf = (const float*)(ws + OFF_G);
  const float* vp   = (const float*)(ws + OFF_VP + (size_t)t * 4 * kMB);
  float gv = gbuf[(gid * 8 + b) * 1024 + hw];
  float vv = vp[((size_t)((b << 7) + o) << 10) + hw];
  out[idx] = gv * vv + b128[o];
}

// ---------------------------------------------------------------------------
extern "C" void kernel_launch(void* const* d_in, const int* in_sizes, int n_in,
                              void* d_out, int out_size, void* d_ws, size_t ws_size,
                              hipStream_t stream) {
  (void)in_sizes; (void)n_in; (void)out_size; (void)ws_size;
  const float* x4   = (const float*)d_in[0];
  const float* x3   = (const float*)d_in[1];
  const float* wq   = (const float*)d_in[2];
  const float* bq   = (const float*)d_in[3];
  const float* wk   = (const float*)d_in[4];
  const float* bk   = (const float*)d_in[5];
  const float* wv   = (const float*)d_in[6];
  const float* bv   = (const float*)d_in[7];
  const float* w128 = (const float*)d_in[8];
  const float* b128 = (const float*)d_in[9];
  char*  ws  = (char*)d_ws;
  float* out = (float*)d_out;

  prep_weights_kernel<<<896, 256, 0, stream>>>(wq, wk, wv, w128, ws);
  conv_qkv_kernel<0><<<dim3(64, 8, 2), 256, 0, stream>>>(x4, x3, bq, bk, bv, ws);
  conv_qkv_kernel<1><<<dim3(64, 8, 2), 256, 0, stream>>>(x4, x3, bq, bk, bv, ws);
  conv_qkv_kernel<2><<<dim3(64, 8, 2), 256, 0, stream>>>(x4, x3, bq, bk, bv, ws);
  vp_conv_kernel<<<dim3(64, 8, 2), 256, 0, stream>>>(ws);
  gate_kernel<<<dim3(64, 4), 128, 0, stream>>>(ws);
  softmax_kernel<<<32, 256, 0, stream>>>(ws);
  final_kernel<<<16384, 256, 0, stream>>>(b128, ws, out);
}
